// MS_69355131896546
// MI455X (gfx1250) — matrix operand order repaired and checked by reading, not compiled, not run on hardware
//
#include <hip/hip_runtime.h>
#include <hip/hip_bf16.h>
#include <math.h>

// ---------------------------------------------------------------------------
// MS motion-statistics op for MI455X (gfx1250, wave32, WMMA).
//   1) norm_transpose : fused per-pixel L2 norm + transpose to pixel-major
//                       bf16 frames nrm[f][p][c] (single pass over x)
//   2) corr_flow      : banded GEMM correlation via v_wmma_f32_16x16x32_bf16
//                       (b128 staging to LDS) + fused silu/argmax/softmax/flow
//   3) dw/pw conv stack with fused SiLU
//   4) pw4 64->512 fused with SiLU + residual add (same flat index) + store
// Roofline: ~300 MB HBM traffic @ 23.3 TB/s => ~13 us floor; compute trivial.
// ---------------------------------------------------------------------------

#define CH   512
#define NT   8
#define NF   16      // 2*8 frames
#define HH   56
#define WW   56
#define HW   3136
#define PAIRS 14

typedef __attribute__((ext_vector_type(16))) __bf16 bf16x16;
typedef __attribute__((ext_vector_type(8)))  float  f32x8;

struct B32 { uint4 a, b; };   // 32-byte chunk for fragment packing

__device__ __forceinline__ float silu_f(float v) {
    return v / (1.0f + __expf(-v));
}

// ---------------------------------------------------------------------------
// Fused normalize + transpose. One block = one frame x 32-pixel tile.
// Read x (b,c,t,h,w) coalesced (pixel fastest), accumulate sum(x^2) per pixel,
// write bf16 normalized tile channel-contiguous: nrm[f][p][c].
// ---------------------------------------------------------------------------
#define TP 514   // LDS row pitch in halves (odd dword pitch -> conflict-free)

__global__ void __launch_bounds__(256)
norm_transpose(const float* __restrict__ x, __bf16* __restrict__ nrm) {
    __shared__ __bf16 T[32][TP];
    __shared__ float  S[32][9];
    __shared__ float  Inv[32];

    const int f  = blockIdx.y;          // 0..15
    const int p0 = blockIdx.x * 32;     // 3136 = 98*32, exact
    const int b  = f >> 3, t = f & 7;
    const int tid = threadIdx.x;
    const int p  = tid & 31;            // pixel within tile
    const int cw = tid >> 5;            // channel group 0..7

    const float* base = x + ((size_t)b * CH * NT + t) * HW + p0 + p;
    float s = 0.0f;
    #pragma unroll 4
    for (int c = cw; c < CH; c += 8) {
        float v = base[(size_t)c * NT * HW];
        s += v * v;
        T[p][c] = (__bf16)v;
    }
    S[p][cw] = s;
    __syncthreads();
    if (tid < 32) {
        float a = 0.0f;
        #pragma unroll
        for (int j = 0; j < 8; ++j) a += S[tid][j];
        Inv[tid] = rsqrtf(a + 1e-6f);
    }
    __syncthreads();

    // write phase: per pixel row, 256 threads emit 256 packed u32 (1 KB)
    unsigned* nrmu = (unsigned*)(nrm + (size_t)f * HW * CH);
    #pragma unroll 4
    for (int pp = 0; pp < 32; ++pp) {
        float iv = Inv[pp];
        float fa = (float)T[pp][2 * tid]     * iv;
        float fb = (float)T[pp][2 * tid + 1] * iv;
        unsigned ua = __builtin_bit_cast(unsigned short, (__bf16)fa);
        unsigned ub = __builtin_bit_cast(unsigned short, (__bf16)fb);
        nrmu[(size_t)(p0 + pp) * (CH / 2) + tid] = (ub << 16) | ua;
    }
}

// ---------------------------------------------------------------------------
// Correlation + flow. One block per (pair, row y). 7 waves, wave w owns dy=w.
// Band GEMM: for x-tile X0 (4 tiles of 16), two 16-wide u-tiles cover
// u in [X0-3, X0+28]; keep the 7-diagonal band. K=512 in 16 steps of 32 (bf16).
// Staging is b128 chunks from pixel-major nrm (channels contiguous).
// ---------------------------------------------------------------------------
#define KSTEP 32
#define BROW  80     // bu = u + 3, bu in [0,79]; valid f2 columns bu-3 in [0,55]

__global__ void __launch_bounds__(224)
corr_flow(const __bf16* __restrict__ nrm, float* __restrict__ x1) {
    __shared__ __align__(16) __bf16 Alds[64 * KSTEP];        //  4 KB
    __shared__ __align__(16) __bf16 Blds[7 * BROW * KSTEP];  // 35 KB
    __shared__ float Cor[WW * 49];                           // 11 KB

    const int pair = blockIdx.x;     // 0..13
    const int y    = blockIdx.y;     // 0..55
    const int b    = pair / 7;
    const int t    = pair - b * 7;
    const int f1   = b * 8 + t;
    const int f2   = f1 + 1;

    const int tid  = threadIdx.x;    // 0..223
    const int wave = tid >> 5;       // 0..6 (7 waves)
    const int lane = tid & 31;

    const __bf16* n1 = nrm + (size_t)f1 * HW * CH;   // pixel-major
    const __bf16* n2 = nrm + (size_t)f2 * HW * CH;

    f32x8 acc[4][2];
    #pragma unroll
    for (int xt = 0; xt < 4; ++xt)
        #pragma unroll
        for (int ut = 0; ut < 2; ++ut)
            #pragma unroll
            for (int r = 0; r < 8; ++r) acc[xt][ut][r] = 0.0f;

    const int kbA = (lane < 16) ? 0 : 8;    // A-frag K base (split halves)
    const int kbB = (lane < 16) ? 0 : 8;    // B-frag K base (split halves)
    const int nn  = lane & 15;

    for (int k0 = 0; k0 < CH; k0 += KSTEP) {
        __syncthreads();
        // --- stage A tile: 64 rows x 4 b128 chunks, zero-pad x>=56
        for (int i = tid; i < 64 * 4; i += 224) {
            int xx = i >> 2, c4 = (i & 3) * 8;
            uint4 v = {0u, 0u, 0u, 0u};
            if (xx < WW)
                v = *(const uint4*)(n1 + ((size_t)(y * WW + xx) * CH + k0 + c4));
            *(uint4*)(Alds + xx * KSTEP + c4) = v;
        }
        // --- stage B tiles: 7*80 rows x 4 b128 chunks, zero-pad OOB
        for (int i = tid; i < 7 * BROW * 4; i += 224) {
            int c4 = (i & 3) * 8;
            int r  = i >> 2;
            int bu = r % BROW;
            int dy = r / BROW;
            int yy = y + dy - 3;
            int uu = bu - 3;
            uint4 v = {0u, 0u, 0u, 0u};
            if (yy >= 0 && yy < HH && uu >= 0 && uu < WW)
                v = *(const uint4*)(n2 + ((size_t)(yy * WW + uu) * CH + k0 + c4));
            *(uint4*)(Blds + (dy * BROW + bu) * KSTEP + c4) = v;
        }
        __syncthreads();

        const int dy = wave;   // wave 0..6
        #pragma unroll
        for (int xt = 0; xt < 4; ++xt) {
            const int X0 = xt * 16;
            // A fragment: lane -> row M = lane%16 (x = X0+M); halves: K split.
            const __bf16* ap = &Alds[(X0 + nn) * KSTEP + kbA];
            B32 ta;
            ta.a = *(const uint4*)(ap);        // k = kbA .. kbA+7
            ta.b = *(const uint4*)(ap + 16);   // k = kbA+16 .. kbA+23
            bf16x16 afrag = __builtin_bit_cast(bf16x16, ta);
            #pragma unroll
            for (int ut = 0; ut < 2; ++ut) {
                const int bu0 = X0 + ut * 16;  // u base = X0-3 (ut=0) / X0+13 (ut=1)
                const __bf16* bp = &Blds[((dy * BROW) + bu0 + nn) * KSTEP + kbB];
                B32 tb;
                tb.a = *(const uint4*)(bp);
                tb.b = *(const uint4*)(bp + 16);
                bf16x16 bfrag = __builtin_bit_cast(bf16x16, tb);
                acc[xt][ut] = __builtin_amdgcn_wmma_f32_16x16x32_bf16(
                    false, afrag, false, bfrag, (short)0, acc[xt][ut],
                    false, false);
            }
        }
    }

    // --- scatter the 7-diagonal band to Cor[x][dy*7 + dx+3]
    {
        const int dy  = wave;
        const int mhi = (lane < 16) ? 0 : 8;
        #pragma unroll
        for (int xt = 0; xt < 4; ++xt) {
            const int X0 = xt * 16;
            #pragma unroll
            for (int ut = 0; ut < 2; ++ut) {
                const int u = X0 + ut * 16 + nn - 3;   // u = bu - 3
                #pragma unroll
                for (int r = 0; r < 8; ++r) {
                    const int xx = X0 + mhi + r;
                    const int dx = u - xx;
                    if (xx < WW && dx >= -3 && dx <= 3)
                        Cor[xx * 49 + dy * 7 + (dx + 3)] = acc[xt][ut][r];
                }
            }
        }
    }
    __syncthreads();

    // --- per-pixel post-processing (silu, conf, argmax, gaussian, softmax, flow)
    if (tid < WW) {
        const int xx = tid;
        float v[49];
        float conf = -3.4e38f;
        int   idx  = 0;
        #pragma unroll
        for (int i = 0; i < 49; ++i) {
            float c = Cor[xx * 49 + i];
            float s = silu_f(c);
            v[i] = s;
            if (s > conf) { conf = s; idx = i; }   // first max (jnp.argmax)
        }
        const float iy = (float)(idx / 7);
        const float ix = (float)(idx % 7);
        float m[49];
        float mmax = -3.4e38f;
        #pragma unroll
        for (int i = 0; i < 7; ++i)
            #pragma unroll
            for (int j = 0; j < 7; ++j) {
                float dyv = (float)i - iy, dxv = (float)j - ix;
                float g = __expf(-(dxv * dxv + dyv * dyv) * (1.0f / 50.0f));
                float mm = g * v[i * 7 + j] * 100.0f;
                m[i * 7 + j] = mm;
                mmax = fmaxf(mmax, mm);
            }
        float se = 0.0f;
        #pragma unroll
        for (int i = 0; i < 49; ++i) { float e = __expf(m[i] - mmax); m[i] = e; se += e; }
        const float inv = 1.0f / se;
        float fx = 0.0f, fy = 0.0f;
        #pragma unroll
        for (int i = 0; i < 7; ++i)
            #pragma unroll
            for (int j = 0; j < 7; ++j) {
                float sm = m[i * 7 + j] * inv;
                fx += sm * (float)(j - 3);
                fy += sm * (float)(i - 3);
            }
        fx *= (1.0f / 3.0f);
        fy *= (1.0f / 3.0f);
        // x1 layout [frame][3][HW]; frame f1 = b*8+t; duplicate t==6 into t==7
        size_t o = ((size_t)f1 * 3) * HW + y * WW + xx;
        x1[o]          = fx;
        x1[o + HW]     = fy;
        x1[o + 2 * HW] = conf;
        if (t == 6) {
            size_t o2 = o + (size_t)3 * HW;
            x1[o2]          = fx;
            x1[o2 + HW]     = fy;
            x1[o2 + 2 * HW] = conf;
        }
    }
}

// ---------------------------------------------------------------------------
// Depthwise conv (KxK, pad P, groups=C) + SiLU. One thread per output element.
// ---------------------------------------------------------------------------
template <int C, int K, int P>
__global__ void dwconv(const float* __restrict__ in, const float* __restrict__ w,
                       float* __restrict__ out) {
    int idx = blockIdx.x * blockDim.x + threadIdx.x;
    if (idx >= NF * C * HW) return;
    int p = idx % HW;
    int r = idx / HW;
    int c = r % C;
    int f = r / C;
    int yy = p / WW, xx = p - yy * WW;
    const float* ip = in + ((size_t)f * C + c) * HW;
    const float* wp = w + c * K * K;
    float s = 0.0f;
    #pragma unroll
    for (int i = 0; i < K; ++i) {
        int ys = yy + i - P;
        if (ys < 0 || ys >= HH) continue;
        #pragma unroll
        for (int j = 0; j < K; ++j) {
            int xs = xx + j - P;
            if (xs < 0 || xs >= WW) continue;
            s += ip[ys * WW + xs] * wp[i * K + j];
        }
    }
    out[idx] = silu_f(s);
}

// ---------------------------------------------------------------------------
// Pointwise conv (CI -> CO) + SiLU. One thread per output element.
// ---------------------------------------------------------------------------
template <int CI, int CO>
__global__ void pwconv(const float* __restrict__ in, const float* __restrict__ w,
                       float* __restrict__ out) {
    int idx = blockIdx.x * blockDim.x + threadIdx.x;
    if (idx >= NF * CO * HW) return;
    int p = idx % HW;
    int r = idx / HW;
    int co = r % CO;
    int f = r / CO;
    const float* ip = in + (size_t)f * CI * HW + p;
    const float* wp = w + co * CI;
    float s = 0.0f;
    #pragma unroll 8
    for (int ci = 0; ci < CI; ++ci) s += ip[(size_t)ci * HW] * wp[ci];
    out[idx] = silu_f(s);
}

// ---------------------------------------------------------------------------
// Final pointwise 64->512 + SiLU + residual add. The reference's final
// .view(b,512,8,h,w) on (16,512,h,w) is a raw reinterpret, so out(f,oc,p)
// and x(b,c,t',p) share the same flat offset -> same-index elementwise add.
// ---------------------------------------------------------------------------
__global__ void pw4_res(const float* __restrict__ in, const float* __restrict__ w,
                        const float* __restrict__ x, float* __restrict__ out) {
    size_t idx = (size_t)blockIdx.x * blockDim.x + threadIdx.x;
    if (idx >= (size_t)NF * 512 * HW) return;
    int p = (int)(idx % HW);
    size_t r = idx / HW;
    int co = (int)(r % 512);
    int f = (int)(r / 512);
    const float* ip = in + (size_t)f * 64 * HW + p;
    const float* wp = w + co * 64;
    float s = 0.0f;
    #pragma unroll 8
    for (int ci = 0; ci < 64; ++ci) s += ip[(size_t)ci * HW] * wp[ci];
    out[idx] = silu_f(s) + x[idx];
}

// ---------------------------------------------------------------------------
extern "C" void kernel_launch(void* const* d_in, const int* in_sizes, int n_in,
                              void* d_out, int out_size, void* d_ws, size_t ws_size,
                              hipStream_t stream) {
    const float* x    = (const float*)d_in[0];
    const float* wdw1 = (const float*)d_in[1];
    const float* wpw1 = (const float*)d_in[2];
    const float* wdw2 = (const float*)d_in[3];
    const float* wpw2 = (const float*)d_in[4];
    const float* wdw3 = (const float*)d_in[5];
    const float* wpw3 = (const float*)d_in[6];
    const float* wdw4 = (const float*)d_in[7];
    const float* wpw4 = (const float*)d_in[8];
    float* out = (float*)d_out;

    char* ws = (char*)d_ws;
    size_t off = 0;
    auto carve = [&](size_t bytes) -> void* {
        void* p = ws + off;
        off += (bytes + 255) & ~(size_t)255;
        return p;
    };
    __bf16* nrm   = (__bf16*)carve((size_t)NF * CH * HW * sizeof(__bf16));   // 51.4 MB
    float*  x1    = (float*) carve((size_t)NF * 3  * HW * sizeof(float));
    float*  tmp3  = (float*) carve((size_t)NF * 3  * HW * sizeof(float));
    float*  act16 = (float*) carve((size_t)NF * 16 * HW * sizeof(float));
    float*  tmp16 = (float*) carve((size_t)NF * 16 * HW * sizeof(float));
    float*  act32 = (float*) carve((size_t)NF * 32 * HW * sizeof(float));
    float*  tmp32 = (float*) carve((size_t)NF * 32 * HW * sizeof(float));
    float*  act64 = (float*) carve((size_t)NF * 64 * HW * sizeof(float));
    float*  tmp64 = (float*) carve((size_t)NF * 64 * HW * sizeof(float));
    (void)ws_size; (void)in_sizes; (void)n_in; (void)out_size;

    const int TB = 256;
    // 1) fused per-pixel L2 norm + transpose to pixel-major bf16
    norm_transpose<<<dim3(HW / 32, NF), 256, 0, stream>>>(x, nrm);
    // 2) WMMA banded-GEMM correlation + fused flow
    corr_flow<<<dim3(PAIRS, HH), 224, 0, stream>>>(nrm, x1);
    // 3) conv stack
    dwconv<3, 5, 2><<<(NF * 3 * HW + TB - 1) / TB, TB, 0, stream>>>(x1, wdw1, tmp3);
    pwconv<3, 16>  <<<(NF * 16 * HW + TB - 1) / TB, TB, 0, stream>>>(tmp3, wpw1, act16);
    dwconv<16, 3, 1><<<(NF * 16 * HW + TB - 1) / TB, TB, 0, stream>>>(act16, wdw2, tmp16);
    pwconv<16, 32> <<<(NF * 32 * HW + TB - 1) / TB, TB, 0, stream>>>(tmp16, wpw2, act32);
    dwconv<32, 3, 1><<<(NF * 32 * HW + TB - 1) / TB, TB, 0, stream>>>(act32, wdw3, tmp32);
    pwconv<32, 64> <<<(NF * 64 * HW + TB - 1) / TB, TB, 0, stream>>>(tmp32, wpw3, act64);
    dwconv<64, 3, 1><<<(NF * 64 * HW + TB - 1) / TB, TB, 0, stream>>>(act64, wdw4, tmp64);
    // 4) pw4 + SiLU + residual + store
    {
        size_t n = (size_t)NF * 512 * HW;
        pw4_res<<<(unsigned)((n + TB - 1) / TB), TB, 0, stream>>>(tmp64, wpw4, x, out);
    }
}
